// ExpertParallel_90357521973821
// MI455X (gfx1250) — matrix operand order repaired and checked by reading, not compiled, not run on hardware
//
#include <hip/hip_runtime.h>

// ---------------------------------------------------------------------------
// MoE expert-parallel forward for MI455X (gfx1250, wave32, WMMA + TDM).
// ~0.55 TFLOP of expert GEMMs via v_wmma_f32_16x16x32_bf16; weights/activations
// pre-converted to bf16 so the GEMM inner loop is copies + WMMA only; B tiles
// staged into padded LDS by the Tensor Data Mover when available.
// ---------------------------------------------------------------------------

#define NTOK 16384
#define DDIM 1024
#define HDIM 4096
#define NEXP 8
#define CAP  8192   // int(1.0 * 16384 / (2*1))

#if defined(__has_builtin)
#if __has_builtin(__builtin_amdgcn_tensor_load_to_lds) && \
    __has_builtin(__builtin_amdgcn_s_wait_tensorcnt)
#define USE_TDM 1
#else
#define USE_TDM 0
#endif
#else
#define USE_TDM 0
#endif

typedef __attribute__((ext_vector_type(4)))  float        f32x4;
typedef __attribute__((ext_vector_type(8)))  float        f32x8;
typedef __attribute__((ext_vector_type(4)))  unsigned int u32x4;
typedef __attribute__((ext_vector_type(16))) __bf16       bf16x16;
typedef __attribute__((ext_vector_type(4)))  unsigned int tdm_g0_t;
typedef __attribute__((ext_vector_type(8)))  int          tdm_g1_t;
typedef __attribute__((ext_vector_type(4)))  int          tdm_g2_t;

union FragAB { bf16x16 v; u32x4 q[2]; };

__device__ inline unsigned short f2bf(float f) {          // f32 -> bf16 RNE
    unsigned int u = __float_as_uint(f);
    u += 0x7FFFu + ((u >> 16) & 1u);
    return (unsigned short)(u >> 16);
}
__device__ inline unsigned int pack2(unsigned short lo, unsigned short hi) {
    return (unsigned int)lo | ((unsigned int)hi << 16);
}
__device__ inline f32x8 wmma_bf16(FragAB a, FragAB b, f32x8 c) {
    return __builtin_amdgcn_wmma_f32_16x16x32_bf16(
        false, a.v, false, b.v, (short)0, c, false, false);
}
__device__ inline unsigned lds_off(const void* p) {       // generic -> LDS byte offset
    return (unsigned)(size_t)p;
}

#if USE_TDM
// TDM: copy a 32(k) x 128(n) bf16 tile, n-major source (row stride = kdim
// elements), into LDS rows of 64B data + 16B pad (Bt[n][40] layout).
__device__ inline void tdm_load_B(unsigned ldsoff, const unsigned short* gsrc,
                                  unsigned kdim) {
    const unsigned long long ga = (unsigned long long)(size_t)gsrc;
    tdm_g0_t g0;
    g0[0] = 1u;                                            // count=1, user mode
    g0[1] = ldsoff;                                        // lds_addr
    g0[2] = (unsigned)ga;                                  // global_addr[31:0]
    g0[3] = (unsigned)((ga >> 32) & 0x1FFFFFFu) | (2u << 30); // addr hi | type=2
    tdm_g1_t g1;
    g1[0] = (int)((1u << 16) | (1u << 20) | (3u << 22) | (3u << 25));
            // data_size=2B | pad_enable | pad_interval=16dw | pad_amount=4dw
    g1[1] = (int)((kdim & 0xFFFFu) << 16);                 // tensor_dim0 lo16
    g1[2] = (int)(((kdim >> 16) & 0xFFFFu) | (128u << 16));// dim0 hi | dim1=128
    g1[3] = (int)(32u << 16);                              // tile_dim0=32
    g1[4] = (int)128;                                      // tile_dim1=128
    g1[5] = (int)kdim;                                     // dim0 stride lo32
    g1[6] = 0;
    g1[7] = 0;
    tdm_g2_t z = {0, 0, 0, 0};
#if defined(__clang_major__) && (__clang_major__ >= 23)
    tdm_g1_t z8 = {0, 0, 0, 0, 0, 0, 0, 0};
    __builtin_amdgcn_tensor_load_to_lds(g0, g1, z, z, z8, 0);
#else
    __builtin_amdgcn_tensor_load_to_lds(g0, g1, z, z, 0);
#endif
}
#endif

// ---------------------------------------------------------------------------
// Router: one wave per token; softmax(E=8); top-2 (strict > == jax tie-break).
// ---------------------------------------------------------------------------
__global__ __launch_bounds__(256) void moe_router(
    const float* __restrict__ x, const float* __restrict__ Wr,
    const float* __restrict__ br, int* __restrict__ tope,
    float* __restrict__ topw, float* __restrict__ pimp)
{
    __shared__ float sProb[8][NEXP];
    const int t = threadIdx.x, wave = t >> 5, lane = t & 31;
    const int n = blockIdx.x * 8 + wave;

    float acc[NEXP];
#pragma unroll
    for (int e = 0; e < NEXP; ++e) acc[e] = 0.f;

    const float* xr = x + (size_t)n * DDIM + lane * 32;
#pragma unroll 4
    for (int i = 0; i < 32; ++i) {
        const float xv = xr[i];
        const int d = lane * 32 + i;
        const f32x4* wp = (const f32x4*)(Wr + (size_t)d * NEXP);
        const f32x4 w0 = wp[0], w1 = wp[1];
#pragma unroll
        for (int j = 0; j < 4; ++j) { acc[j] += xv * w0[j]; acc[4 + j] += xv * w1[j]; }
    }
#pragma unroll
    for (int e = 0; e < NEXP; ++e)
        for (int off = 16; off; off >>= 1)
            acc[e] += __shfl_xor(acc[e], off, 32);

    float lg[NEXP], p[NEXP];
    float mx = -1e30f;
#pragma unroll
    for (int e = 0; e < NEXP; ++e) { lg[e] = acc[e] + br[e]; mx = fmaxf(mx, lg[e]); }
    float s = 0.f;
#pragma unroll
    for (int e = 0; e < NEXP; ++e) { p[e] = __expf(lg[e] - mx); s += p[e]; }
    const float inv = 1.f / s;
#pragma unroll
    for (int e = 0; e < NEXP; ++e) p[e] *= inv;

    if (lane == 0) {
        int e0 = 0; float p0 = p[0];
#pragma unroll
        for (int e = 1; e < NEXP; ++e) if (p[e] > p0) { p0 = p[e]; e0 = e; }
        int e1 = -1; float p1 = -1.f;
#pragma unroll
        for (int e = 0; e < NEXP; ++e) if (e != e0 && p[e] > p1) { p1 = p[e]; e1 = e; }
        tope[2 * n]     = e0;  topw[2 * n]     = p0;
        tope[2 * n + 1] = e1;  topw[2 * n + 1] = p1;
#pragma unroll
        for (int e = 0; e < NEXP; ++e) sProb[wave][e] = p[e];
    }
    __syncthreads();
    if (t < NEXP) {
        float si = 0.f;
#pragma unroll
        for (int w = 0; w < 8; ++w) si += sProb[w][t];
        pimp[(size_t)blockIdx.x * NEXP + t] = si;
    }
}

// ---------------------------------------------------------------------------
// Capacity scan: one block per expert; token-order prefix sum.
// ---------------------------------------------------------------------------
__global__ __launch_bounds__(1024) void moe_scan(
    const int* __restrict__ tope, const float* __restrict__ topw,
    int* __restrict__ sel, float* __restrict__ gatew,
    int* __restrict__ kept, float* __restrict__ cntf)
{
    __shared__ int sdata[1024];
    const int e = blockIdx.x, t = threadIdx.x;
    int running = 0;
    for (int chunk = 0; chunk < NTOK / 1024; ++chunk) {
        const int n = chunk * 1024 + t;
        const int e0 = tope[2 * n], e1 = tope[2 * n + 1];
        const int m = (e0 == e) || (e1 == e);
        const float w = (e0 == e) ? topw[2 * n] : ((e1 == e) ? topw[2 * n + 1] : 0.f);
        sdata[t] = m;
        __syncthreads();
        for (int off = 1; off < 1024; off <<= 1) {
            const int y = (t >= off) ? sdata[t - off] : 0;
            __syncthreads();
            sdata[t] += y;
            __syncthreads();
        }
        const int incl = sdata[t];
        const int total = sdata[1023];
        const int pos = running + incl - 1;
        if (m && pos < CAP) {
            sel[(size_t)e * CAP + pos] = n;
            gatew[(size_t)e * CAP + pos] = w;
        }
        running += total;
        __syncthreads();
    }
    if (t == 0) {
        cntf[e] = (float)running;
        kept[e] = running < CAP ? running : CAP;
    }
}

// ---------------------------------------------------------------------------
__global__ __launch_bounds__(32) void moe_aux(
    const float* __restrict__ pimp, const float* __restrict__ cntf,
    float* __restrict__ outaux)
{
    __shared__ float sImp[NEXP];
    const int t = threadIdx.x;
    if (t < NEXP) {
        float si = 0.f;
        for (int b = 0; b < NTOK / 8; ++b) si += pimp[(size_t)b * NEXP + t];
        sImp[t] = si;
    }
    __syncthreads();
    if (t == 0) {
        float bal = 0.f, il = 0.f;
        for (int e = 0; e < NEXP; ++e) {
            const float imp = sImp[e], cnt = cntf[e];
            bal += (imp / NTOK) * (cnt / NTOK);
            il  += imp * imp;
        }
        outaux[0] = bal * NEXP + il / NEXP;
    }
}

// ---------------------------------------------------------------------------
// Elementwise f32 -> bf16 (for x), 8 values/thread.
// ---------------------------------------------------------------------------
__global__ __launch_bounds__(256) void cvt_bf16(
    const float* __restrict__ in, unsigned short* __restrict__ out)
{
    const size_t i = ((size_t)blockIdx.x * 256 + threadIdx.x) * 8;
    const f32x4 a = *(const f32x4*)(in + i);
    const f32x4 b = *(const f32x4*)(in + i + 4);
    u32x4 o;
    o[0] = pack2(f2bf(a[0]), f2bf(a[1]));
    o[1] = pack2(f2bf(a[2]), f2bf(a[3]));
    o[2] = pack2(f2bf(b[0]), f2bf(b[1]));
    o[3] = pack2(f2bf(b[2]), f2bf(b[3]));
    *(u32x4*)(out + i) = o;
}

// ---------------------------------------------------------------------------
// Transpose+convert: in f32 [R][C] -> out bf16 [C][R] (weights to n-major).
// ---------------------------------------------------------------------------
__global__ __launch_bounds__(256) void transpose_cvt(
    const float* __restrict__ in, unsigned short* __restrict__ out,
    int R, int C)
{
    __shared__ float tile[32][33];
    const int tx = threadIdx.x & 31, ty = threadIdx.x >> 5;
    const int c0 = blockIdx.x * 32, r0 = blockIdx.y * 32;
#pragma unroll
    for (int i = 0; i < 4; ++i)
        tile[ty + 8 * i][tx] = in[(size_t)(r0 + ty + 8 * i) * C + c0 + tx];
    __syncthreads();
#pragma unroll
    for (int i = 0; i < 4; ++i)
        out[(size_t)(c0 + ty + 8 * i) * R + r0 + tx] = f2bf(tile[tx][ty + 8 * i]);
}

// ---------------------------------------------------------------------------
// GEMM1: h = relu(gather(xbf) @ W1 + b1) -> bf16 hbuf.
// 8 waves (4x2), BM=64 BN=128 BK=32, pointer-swapped double-buffered LDS
// (single loop body -> in-place WMMA accumulators), TDM B staging.
// ---------------------------------------------------------------------------
__global__ __launch_bounds__(256) void moe_gemm1(
    const unsigned short* __restrict__ xbf, const unsigned short* __restrict__ w1t,
    const float* __restrict__ b1e, const int* __restrict__ sel_e,
    const int* __restrict__ keptp, unsigned short* __restrict__ hb)
{
    __shared__ unsigned short As[2][64 * 40];
    __shared__ unsigned short Bs[2][128 * 40];
    const int kept = *keptp;
    const int c0 = blockIdx.x * 64;
    if (c0 >= kept) return;                                // block-uniform
    const int j0 = blockIdx.y * 128;

    const int t = threadIdx.x, lane = t & 31, wave = t >> 5;
    const int wm = wave & 3, wn = wave >> 2;
    const int arow = t >> 2, asub = t & 3;
    const int half = lane >> 4, nlo = lane & 15;
    const int brow = t >> 2, bsub = t & 3;                 // manual-B fallback

    const unsigned short* xr = nullptr;
    {
        const int slot = c0 + arow;
        if (slot < kept) xr = xbf + (size_t)sel_e[slot] * DDIM + asub * 8;
    }
    const unsigned short* bsrc = w1t + (size_t)j0 * DDIM;  // [H][D] n-major

    const int aoff = (wm * 16 + nlo) * 80 + half * 32;
    const int astore = arow * 80 + asub * 16;
    const int nb = (wn * 64 + nlo) * 80 + half * 32;
    const u32x4 zq = {0, 0, 0, 0};

    // ---- prologue: stage tile kb=0 into buffer 0 ----
    {
        u32x4 aq = zq;
        if (xr) aq = *(const u32x4*)xr;
        *(u32x4*)((char*)As[0] + astore) = aq;
#if USE_TDM
        if (wave == 0) tdm_load_B(lds_off(Bs[0]), bsrc, DDIM);
        if (wave == 0) __builtin_amdgcn_s_wait_tensorcnt(0);
#else
        const u32x4 b0 = *(const u32x4*)(bsrc + (size_t)brow * DDIM + bsub * 8);
        const u32x4 b1 = *(const u32x4*)(bsrc + (size_t)(64 + brow) * DDIM + bsub * 8);
        *(u32x4*)((char*)Bs[0] + brow * 80 + bsub * 16) = b0;
        *(u32x4*)((char*)Bs[0] + (64 + brow) * 80 + bsub * 16) = b1;
#endif
    }
    __syncthreads();

    f32x8 acc0 = {}, acc1 = {}, acc2 = {}, acc3 = {};
    const char* curA = (const char*)As[0];
    const char* curB = (const char*)Bs[0];
    char* nxtA = (char*)As[1];
    char* nxtB = (char*)Bs[1];

    const int KT = DDIM / 32;
#pragma clang loop unroll(disable)
    for (int kb = 0; kb < KT; ++kb) {
        const bool more = (kb + 1 < KT);
        u32x4 aq = zq, bq0 = zq, bq1 = zq;
        if (more) {
            if (xr) aq = *(const u32x4*)(xr + (kb + 1) * 32);
#if USE_TDM
            if (wave == 0) tdm_load_B(lds_off(nxtB), bsrc + (kb + 1) * 32, DDIM);
#else
            bq0 = *(const u32x4*)(bsrc + (size_t)brow * DDIM + (kb + 1) * 32 + bsub * 8);
            bq1 = *(const u32x4*)(bsrc + (size_t)(64 + brow) * DDIM + (kb + 1) * 32 + bsub * 8);
#endif
        }
        // ---- compute current tile: cluster frag loads, 4 back-to-back WMMAs ----
        FragAB a, b0, b1, b2, b3;
        a.q[0]  = *(const u32x4*)(curA + aoff);
        a.q[1]  = *(const u32x4*)(curA + aoff + 16);
        b0.q[0] = *(const u32x4*)(curB + nb);
        b0.q[1] = *(const u32x4*)(curB + nb + 16);
        b1.q[0] = *(const u32x4*)(curB + nb + 16 * 80);
        b1.q[1] = *(const u32x4*)(curB + nb + 16 * 80 + 16);
        b2.q[0] = *(const u32x4*)(curB + nb + 32 * 80);
        b2.q[1] = *(const u32x4*)(curB + nb + 32 * 80 + 16);
        b3.q[0] = *(const u32x4*)(curB + nb + 48 * 80);
        b3.q[1] = *(const u32x4*)(curB + nb + 48 * 80 + 16);
        acc0 = wmma_bf16(a, b0, acc0);
        acc1 = wmma_bf16(a, b1, acc1);
        acc2 = wmma_bf16(a, b2, acc2);
        acc3 = wmma_bf16(a, b3, acc3);

        if (more) {
            *(u32x4*)(nxtA + astore) = aq;
#if !USE_TDM
            *(u32x4*)(nxtB + brow * 80 + bsub * 16) = bq0;
            *(u32x4*)(nxtB + (64 + brow) * 80 + bsub * 16) = bq1;
#endif
        }
#if USE_TDM
        if (wave == 0) __builtin_amdgcn_s_wait_tensorcnt(0);
#endif
        __syncthreads();
        // swap double buffers
        const char* tA = curA; curA = (const char*)nxtA; nxtA = (char*)tA;
        const char* tB = curB; curB = (const char*)nxtB; nxtB = (char*)tB;
    }

    // ---- epilogue: +b1, ReLU, bf16 store ----
    const int rbase = half * 8;
    f32x8 accs[4] = { acc0, acc1, acc2, acc3 };
#pragma unroll
    for (int t4 = 0; t4 < 4; ++t4) {
        const int col = j0 + wn * 64 + t4 * 16 + nlo;
        const float bv = b1e[col];
#pragma unroll
        for (int r = 0; r < 8; ++r) {
            const int c = c0 + wm * 16 + rbase + r;
            float v = accs[t4][r] + bv;
            v = v > 0.f ? v : 0.f;
            hb[(size_t)c * HDIM + col] = f2bf(v);
        }
    }
}

// ---------------------------------------------------------------------------
// GEMM2: out = h @ W2 + b2, * gate, non-atomic RMW scatter into final
// (each token appears once per expert; expert launches serialize).
// ---------------------------------------------------------------------------
__global__ __launch_bounds__(256) void moe_gemm2(
    const unsigned short* __restrict__ hb, const unsigned short* __restrict__ w2t,
    const float* __restrict__ b2e, const int* __restrict__ sel_e,
    const float* __restrict__ gw_e, const int* __restrict__ keptp,
    float* __restrict__ outf)
{
    __shared__ unsigned short As[2][64 * 40];
    __shared__ unsigned short Bs[2][128 * 40];
    __shared__ int   sTok[64];
    __shared__ float sW[64];
    const int kept = *keptp;
    const int c0 = blockIdx.x * 64;
    if (c0 >= kept) return;
    const int d0 = blockIdx.y * 128;

    const int t = threadIdx.x, lane = t & 31, wave = t >> 5;
    const int wm = wave & 3, wn = wave >> 2;
    const int arow = t >> 2, asub = t & 3;
    const int half = lane >> 4, nlo = lane & 15;
    const int brow = t >> 2, bsub = t & 3;

    if (t < 64) {
        const int slot = c0 + t;
        if (slot < kept) { sTok[t] = sel_e[slot]; sW[t] = gw_e[slot]; }
        else             { sTok[t] = -1;          sW[t] = 0.f; }
    }

    const unsigned short* hr   = hb + (size_t)(c0 + arow) * HDIM + asub * 8;
    const unsigned short* bsrc = w2t + (size_t)d0 * HDIM;  // [D][H] n-major

    const int aoff = (wm * 16 + nlo) * 80 + half * 32;
    const int astore = arow * 80 + asub * 16;
    const int nb = (wn * 64 + nlo) * 80 + half * 32;
    const u32x4 zq = {0, 0, 0, 0};

    {
        const u32x4 aq = *(const u32x4*)hr;
        *(u32x4*)((char*)As[0] + astore) = aq;
#if USE_TDM
        if (wave == 0) tdm_load_B(lds_off(Bs[0]), bsrc, HDIM);
        if (wave == 0) __builtin_amdgcn_s_wait_tensorcnt(0);
#else
        const u32x4 b0 = *(const u32x4*)(bsrc + (size_t)brow * HDIM + bsub * 8);
        const u32x4 b1 = *(const u32x4*)(bsrc + (size_t)(64 + brow) * HDIM + bsub * 8);
        *(u32x4*)((char*)Bs[0] + brow * 80 + bsub * 16) = b0;
        *(u32x4*)((char*)Bs[0] + (64 + brow) * 80 + bsub * 16) = b1;
#endif
    }
    __syncthreads();

    f32x8 acc0 = {}, acc1 = {}, acc2 = {}, acc3 = {};
    const char* curA = (const char*)As[0];
    const char* curB = (const char*)Bs[0];
    char* nxtA = (char*)As[1];
    char* nxtB = (char*)Bs[1];

    const int KT = HDIM / 32;
#pragma clang loop unroll(disable)
    for (int kb = 0; kb < KT; ++kb) {
        const bool more = (kb + 1 < KT);
        u32x4 aq = zq, bq0 = zq, bq1 = zq;
        if (more) {
            aq = *(const u32x4*)(hr + (kb + 1) * 32);
#if USE_TDM
            if (wave == 0) tdm_load_B(lds_off(nxtB), bsrc + (kb + 1) * 32, HDIM);
#else
            bq0 = *(const u32x4*)(bsrc + (size_t)brow * HDIM + (kb + 1) * 32 + bsub * 8);
            bq1 = *(const u32x4*)(bsrc + (size_t)(64 + brow) * HDIM + (kb + 1) * 32 + bsub * 8);
#endif
        }
        FragAB a, b0, b1, b2, b3;
        a.q[0]  = *(const u32x4*)(curA + aoff);
        a.q[1]  = *(const u32x4*)(curA + aoff + 16);
        b0.q[0] = *(const u32x4*)(curB + nb);
        b0.q[1] = *(const u32x4*)(curB + nb + 16);
        b1.q[0] = *(const u32x4*)(curB + nb + 16 * 80);
        b1.q[1] = *(const u32x4*)(curB + nb + 16 * 80 + 16);
        b2.q[0] = *(const u32x4*)(curB + nb + 32 * 80);
        b2.q[1] = *(const u32x4*)(curB + nb + 32 * 80 + 16);
        b3.q[0] = *(const u32x4*)(curB + nb + 48 * 80);
        b3.q[1] = *(const u32x4*)(curB + nb + 48 * 80 + 16);
        acc0 = wmma_bf16(a, b0, acc0);
        acc1 = wmma_bf16(a, b1, acc1);
        acc2 = wmma_bf16(a, b2, acc2);
        acc3 = wmma_bf16(a, b3, acc3);

        if (more) {
            *(u32x4*)(nxtA + astore) = aq;
#if !USE_TDM
            *(u32x4*)(nxtB + brow * 80 + bsub * 16) = bq0;
            *(u32x4*)(nxtB + (64 + brow) * 80 + bsub * 16) = bq1;
#endif
        }
#if USE_TDM
        if (wave == 0) __builtin_amdgcn_s_wait_tensorcnt(0);
#endif
        __syncthreads();
        const char* tA = curA; curA = (const char*)nxtA; nxtA = (char*)tA;
        const char* tB = curB; curB = (const char*)nxtB; nxtB = (char*)tB;
    }

    const int rbase = half * 8;
    f32x8 accs[4] = { acc0, acc1, acc2, acc3 };
#pragma unroll
    for (int t4 = 0; t4 < 4; ++t4) {
        const int col = d0 + wn * 64 + t4 * 16 + nlo;
        const float bv = b2e[col];
#pragma unroll
        for (int r = 0; r < 8; ++r) {
            const int cl = wm * 16 + rbase + r;
            const int tok = sTok[cl];
            if (tok >= 0) {
                float* p = outf + (size_t)tok * DDIM + col;
                *p += (accs[t4][r] + bv) * sW[cl];
            }
        }
    }
}

// ---------------------------------------------------------------------------
extern "C" void kernel_launch(void* const* d_in, const int* in_sizes, int n_in,
                              void* d_out, int out_size, void* d_ws, size_t ws_size,
                              hipStream_t stream) {
    (void)in_sizes; (void)n_in; (void)ws_size;
    const float* x  = (const float*)d_in[0];
    const float* Wr = (const float*)d_in[1];
    const float* br = (const float*)d_in[2];
    const float* W1 = (const float*)d_in[3];
    const float* b1 = (const float*)d_in[4];
    const float* W2 = (const float*)d_in[5];
    const float* b2 = (const float*)d_in[6];
    float* finalOut = (float*)d_out;

    // workspace layout (256B aligned slices)
    char* ws = (char*)d_ws;
    float* pimp  = (float*)(ws + 0);                       //  64 KB
    int*   tope  = (int*)  (ws + 65536);                   // 128 KB
    float* topw  = (float*)(ws + 196608);                  // 128 KB
    int*   sel   = (int*)  (ws + 327680);                  // 256 KB
    float* gatew = (float*)(ws + 589824);                  // 256 KB
    int*   kept  = (int*)  (ws + 851968);
    float* cntf  = (float*)(ws + 852224);
    unsigned short* xbf  = (unsigned short*)(ws + 852480);           // 32 MB
    unsigned short* w1t  = (unsigned short*)(ws + 852480 + 33554432);//  8 MB
    unsigned short* w2t  = (unsigned short*)(ws + 852480 + 41943040);//  8 MB
    unsigned short* hbuf = (unsigned short*)(ws + 852480 + 50331648);// 64 MB

    hipMemsetAsync(d_out, 0, (size_t)out_size * sizeof(float), stream);

    cvt_bf16<<<(NTOK * DDIM) / 2048, 256, 0, stream>>>(x, xbf);
    moe_router<<<NTOK / 8, 256, 0, stream>>>(x, Wr, br, tope, topw, pimp);
    moe_scan<<<NEXP, 1024, 0, stream>>>(tope, topw, sel, gatew, kept, cntf);
    moe_aux<<<1, 32, 0, stream>>>(pimp, cntf, finalOut + (size_t)NTOK * DDIM);

    for (int e = 0; e < NEXP; ++e) {
        transpose_cvt<<<dim3(HDIM / 32, DDIM / 32), 256, 0, stream>>>(
            W1 + (size_t)e * DDIM * HDIM, w1t, DDIM, HDIM);
        transpose_cvt<<<dim3(DDIM / 32, HDIM / 32), 256, 0, stream>>>(
            W2 + (size_t)e * HDIM * DDIM, w2t, HDIM, DDIM);
        moe_gemm1<<<dim3(CAP / 64, HDIM / 128), 256, 0, stream>>>(
            xbf, w1t, b1 + (size_t)e * HDIM, sel + (size_t)e * CAP, kept + e, hbuf);
        moe_gemm2<<<dim3(CAP / 64, DDIM / 128), 256, 0, stream>>>(
            hbuf, w2t, b2 + (size_t)e * DDIM, sel + (size_t)e * CAP,
            gatew + (size_t)e * CAP, kept + e, finalOut);
    }
}
